// GCNLayer_29308856828249
// MI455X (gfx1250) — matrix operand order repaired and checked by reading, not compiled, not run on hardware
//
#include <hip/hip_runtime.h>
#include <hip/hip_bf16.h>

#define NN     8192
#define DIN    128
#define DTH    256
#define DH     64      // D_OUT/2
#define LOG2E  1.44269504088896340736f

typedef __attribute__((ext_vector_type(16))) __bf16        v16bf;
typedef __attribute__((ext_vector_type(8)))  float         v8f;
typedef __attribute__((ext_vector_type(4)))  unsigned int  u32x4;

union V16 { u32x4 u[2]; v16bf b; };

__device__ __forceinline__ unsigned short f2bf(float f) {
    unsigned u = __float_as_uint(f);
    unsigned r = (u + 0x7fffu + ((u >> 16) & 1u)) >> 16;   // RNE
    return (unsigned short)r;
}

// ---------------------------------------------------------------------------
// Kernel 1: BatchNorm statistics -> per-column affine (scale, shift)
// ---------------------------------------------------------------------------
__global__ void bn_stats_kernel(const float* __restrict__ H,
                                const float* __restrict__ gamma,
                                const float* __restrict__ beta,
                                float* __restrict__ scale,
                                float* __restrict__ shift) {
    const int c = blockIdx.x;           // column 0..127
    const int t = threadIdx.x;          // 256 threads
    float s = 0.f, ss = 0.f;
    for (int r = t; r < NN; r += 256) {
        float v = H[(size_t)r * DIN + c];
        s += v; ss += v * v;
    }
    __shared__ float sh[256], sh2[256];
    sh[t] = s; sh2[t] = ss;
    __syncthreads();
    for (int o = 128; o > 0; o >>= 1) {
        if (t < o) { sh[t] += sh[t + o]; sh2[t] += sh2[t + o]; }
        __syncthreads();
    }
    if (t == 0) {
        float mean = sh[0] * (1.f / NN);
        float var  = sh2[0] * (1.f / NN) - mean * mean;       // biased (jnp.var)
        float sc   = rsqrtf(var + 1e-5f) * gamma[c];
        scale[c] = sc;
        shift[c] = beta[c] - mean * sc;
    }
}

// ---------------------------------------------------------------------------
// Kernel 2: pack adjacency A (0.0/1.0 fp32) into a bitmask (8MB, L2 resident)
// ---------------------------------------------------------------------------
__global__ void maskpack_kernel(const float* __restrict__ A,
                                unsigned* __restrict__ mw) {
    const int row  = blockIdx.x;
    const int wid  = threadIdx.x >> 5;
    const int lane = threadIdx.x & 31;
    const float* ap = A + (size_t)row * NN + wid * 1024;
    unsigned myw = 0;
#pragma unroll
    for (int k = 0; k < 32; ++k) {
        float a = ap[k * 32 + lane];
        unsigned long long b = __ballot(a > 0.f);
        if (lane == k) myw = (unsigned)b;
    }
    mw[(size_t)row * 256 + wid * 32 + lane] = myw;
}

// ---------------------------------------------------------------------------
// Kernel 3: prep — 16 rows per block: Hn -> Hx bf16 [N,256],
// out1 -> d_out cols 0..63, B2^T bf16 [64,N]
// ---------------------------------------------------------------------------
__global__ void prep_kernel(const float* __restrict__ H,
                            const float* __restrict__ scale,
                            const float* __restrict__ shift,
                            const float* __restrict__ Wt,
                            const float* __restrict__ bt,
                            const float* __restrict__ W1,
                            const float* __restrict__ b1,
                            const float* __restrict__ W2,
                            const float* __restrict__ b2,
                            unsigned short* __restrict__ Hxb,
                            unsigned short* __restrict__ BT,
                            float* __restrict__ out) {
    __shared__ float hn[16][DIN];
    const int t  = threadIdx.x;
    const int r0 = blockIdx.x * 16;

#pragma unroll
    for (int e = 0; e < 8; ++e) {
        int idx = t + e * 256;
        int r = idx >> 7, c = idx & 127;
        hn[r][c] = H[(size_t)(r0 + r) * DIN + c] * scale[c] + shift[c];
    }
    __syncthreads();

    {   // Hx column t for all 16 rows
        float acc[16];
        float bv = bt[t];
#pragma unroll
        for (int r = 0; r < 16; ++r) acc[r] = bv;
        for (int k = 0; k < DIN; ++k) {
            float w = Wt[(size_t)k * DTH + t];
#pragma unroll
            for (int r = 0; r < 16; ++r) acc[r] += hn[r][k] * w;
        }
#pragma unroll
        for (int r = 0; r < 16; ++r)
            Hxb[(size_t)(r0 + r) * DTH + t] = f2bf(acc[r]);
    }

    {   // out1 (cols 0..63) and B2 (cols 0..63), 8 rows per thread
        const int half = t >> 7;
        const int cc   = t & 127;
        const int c    = cc & 63;
        const float* W = (cc < DH) ? W1 : W2;
        float bv = (cc < DH) ? b1[c] : b2[c];
        float acc[8];
#pragma unroll
        for (int r = 0; r < 8; ++r) acc[r] = bv;
        for (int k = 0; k < DIN; ++k) {
            float w = W[(size_t)k * DH + c];
#pragma unroll
            for (int r = 0; r < 8; ++r) acc[r] += hn[half * 8 + r][k] * w;
        }
        if (cc < DH) {
#pragma unroll
            for (int r = 0; r < 8; ++r) {
                float v = acc[r];
                v = v > 0.f ? v : 0.01f * v;                  // LeakyReLU
                out[(size_t)(r0 + half * 8 + r) * 128 + cc] = v;
            }
        } else {
#pragma unroll
            for (int r = 0; r < 8; ++r)
                BT[(size_t)c * NN + (r0 + half * 8 + r)] = f2bf(acc[r]);
        }
    }
}

// ---------------------------------------------------------------------------
// Async global->LDS staging of one 32-column tile (shared by all 8 waves):
//   Hx[jb..jb+31, 0..255]  : contiguous 16KB  -> hxdst
//   B2^T[0..63, jb..jb+31] : 64 rows x 64B    -> btdst (row-major 64x32)
// Each GLOBAL_LOAD_ASYNC_TO_LDS_B128 moves 16B/lane = 512B/wave.
// ---------------------------------------------------------------------------
__device__ __forceinline__ void stage_tiles(const unsigned short* __restrict__ Hxb,
                                            const unsigned short* __restrict__ BT,
                                            unsigned short* hxdst,
                                            unsigned short* btdst,
                                            int jb, int t) {
    const unsigned short* src = Hxb + (size_t)jb * DTH;     // 16KB contiguous
#pragma unroll
    for (int q = 0; q < 4; ++q) {
        int f = t + q * 256;                                 // 0..1023 chunks
        unsigned lds_off = (unsigned)(uintptr_t)(hxdst + f * 8);
        unsigned long long ga = (unsigned long long)(uintptr_t)(src + f * 8);
        asm volatile("global_load_async_to_lds_b128 %0, %1, off"
                     :: "v"(lds_off), "v"(ga) : "memory");
    }
    {
        int row = t >> 2, chunk = t & 3;                     // 64 rows x 4 chunks
        unsigned lds_off = (unsigned)(uintptr_t)(btdst + t * 8);
        unsigned long long ga =
            (unsigned long long)(uintptr_t)(BT + (size_t)row * NN + jb + chunk * 8);
        asm volatile("global_load_async_to_lds_b128 %0, %1, off"
                     :: "v"(lds_off), "v"(ga) : "memory");
    }
}

// ---------------------------------------------------------------------------
// Kernel 4: fused  sigmoid(Hx Hx^T) -> mask -> softmax -> @ B2 -> leaky.
// One wave per 16-row tile; j-tiles double-buffered through LDS via async DMA.
// Row-max is exactly 1.0 (forced diagonal) => no online-max rescaling.
// ---------------------------------------------------------------------------
__global__ void __launch_bounds__(256, 1)
fused_attn_kernel(const unsigned short* __restrict__ Hxb,
                  const unsigned short* __restrict__ BT,
                  const unsigned* __restrict__ mw,
                  float* __restrict__ out) {
    __shared__ __align__(16) unsigned short pbuf[8][16 * 32];   // 8KB  P staging
    __shared__ __align__(16) unsigned short hxs[2][32 * DTH];   // 32KB Hx tiles
    __shared__ __align__(16) unsigned short bts[2][64 * 32];    // 8KB  B2^T tiles

    const int t    = threadIdx.x;
    const int wid  = t >> 5;
    const int lane = t & 31;
    const int half = lane >> 4;        // lane half 0|1
    const int ln   = lane & 15;
    const int i    = (blockIdx.x * 8 + wid) * 16;   // my 16 rows

    // --- preload A-operand: Hx rows i..i+15, K=0..255 (held in regs) -------
    // A layout: lanes0-15 row=ln hold K {0-7,16-23}; lanes16-31 {8-15,24-31}
    V16 aop[8];
    {
        const unsigned short* base = Hxb + (size_t)(i + ln) * DTH + half * 8;
#pragma unroll
        for (int kb = 0; kb < 8; ++kb) {
            aop[kb].u[0] = *(const u32x4*)(base + kb * 32);
            aop[kb].u[1] = *(const u32x4*)(base + kb * 32 + 16);
        }
    }

    v8f acc[4];
#pragma unroll
    for (int tt = 0; tt < 4; ++tt) acc[tt] = v8f{0.f,0.f,0.f,0.f,0.f,0.f,0.f,0.f};
    float zp[8];
#pragma unroll
    for (int r = 0; r < 8; ++r) zp[r] = 0.f;

    // prologue: stage first tile
    stage_tiles(Hxb, BT, hxs[0], bts[0], 0, t);
    asm volatile("s_wait_asynccnt 0x0" ::: "memory");
    __syncthreads();

    int buf = 0;
    for (int jb = 0; jb < NN; jb += 32) {
        if (jb + 32 < NN)   // prefetch next tile into the other buffer
            stage_tiles(Hxb, BT, hxs[buf ^ 1], bts[buf ^ 1], jb + 32, t);

        // mask words for my 16 rows (one load/lane, broadcast via shuffle)
        unsigned wrow = mw[(size_t)(i + ln) * 256 + (jb >> 5)];

#pragma unroll
        for (int hj = 0; hj < 2; ++hj) {
            const int j = jb + hj * 16;
            v8f c8 = v8f{0.f,0.f,0.f,0.f,0.f,0.f,0.f,0.f};
            // B layout: lanes0-15 col=ln hold K {0-7,16-23}; lanes16-31 {8-15,24-31}
            const unsigned short* bb0 = &hxs[buf][(hj * 16 + ln) * DTH + half * 8];
#pragma unroll
            for (int kb = 0; kb < 8; ++kb) {
                V16 bop;
                bop.u[0] = *(const u32x4*)(bb0 + kb * 32);
                bop.u[1] = *(const u32x4*)(bb0 + kb * 32 + 16);
                c8 = __builtin_amdgcn_wmma_f32_16x16x32_bf16(
                        false, aop[kb].b, false, bop.b, (short)0, c8, false, false);
            }
            // postprocess: mask, sigmoid, exp (row max is the const diagonal 1.0)
            const int col = j + ln;
#pragma unroll
            for (int r = 0; r < 8; ++r) {
                const int row = i + r + 8 * half;              // C/D layout M
                unsigned w = __shfl(wrow, r + 8 * half, 32);
                bool on = (w >> (col & 31)) & 1u;
                float s  = c8[r];
                float sg = __builtin_amdgcn_rcpf(1.f + __builtin_amdgcn_exp2f(-s * LOG2E));
                float p  = on ? __builtin_amdgcn_exp2f(sg * LOG2E) : 0.f;
                if (col == row) p = 2.71828182845904523536f;   // exp(1.0) diagonal
                zp[r] += p;
                pbuf[wid][(r + 8 * half) * 32 + hj * 16 + ln] = f2bf(p);
            }
        }
        // same-wave DS ordering for the P tile
        asm volatile("s_wait_dscnt 0x0" ::: "memory");

        // re-read P tile (16x32) in A-operand layout
        V16 pop;
        {
            const unsigned short* pb = &pbuf[wid][ln * 32 + half * 8];
            pop.u[0] = *(const u32x4*)(pb);
            pop.u[1] = *(const u32x4*)(pb + 16);
        }
        asm volatile("" ::: "memory");

        // out2 tile += P(16x32) @ B2[jb..jb+31, 0..63]
#pragma unroll
        for (int tt = 0; tt < 4; ++tt) {
            V16 bop;
            const unsigned short* bb = &bts[buf][(tt * 16 + ln) * 32 + half * 8];
            bop.u[0] = *(const u32x4*)(bb);
            bop.u[1] = *(const u32x4*)(bb + 16);
            acc[tt] = __builtin_amdgcn_wmma_f32_16x16x32_bf16(
                        false, pop.b, false, bop.b, (short)0, acc[tt], false, false);
        }

        // next tile staged + everyone done reading current buffers
        asm volatile("s_wait_asynccnt 0x0" ::: "memory");
        __syncthreads();
        buf ^= 1;
    }

    // reduce Z across the 16 lanes of each half (xor masks stay in-half)
    float z[8];
#pragma unroll
    for (int r = 0; r < 8; ++r) {
        float v = zp[r];
        v += __shfl_xor(v, 1, 32);
        v += __shfl_xor(v, 2, 32);
        v += __shfl_xor(v, 4, 32);
        v += __shfl_xor(v, 8, 32);
        z[r] = __builtin_amdgcn_rcpf(v);
    }

    // normalize, LeakyReLU, store to d_out columns 64..127
#pragma unroll
    for (int tt = 0; tt < 4; ++tt) {
#pragma unroll
        for (int r = 0; r < 8; ++r) {
            float v = acc[tt][r] * z[r];
            v = v > 0.f ? v : 0.01f * v;
            out[(size_t)(i + r + 8 * half) * 128 + 64 + tt * 16 + ln] = v;
        }
    }
}

// ---------------------------------------------------------------------------
extern "C" void kernel_launch(void* const* d_in, const int* in_sizes, int n_in,
                              void* d_out, int out_size, void* d_ws, size_t ws_size,
                              hipStream_t stream) {
    const float* H     = (const float*)d_in[0];
    const float* A     = (const float*)d_in[1];
    const float* gamma = (const float*)d_in[2];
    const float* beta  = (const float*)d_in[3];
    const float* Wt    = (const float*)d_in[4];
    const float* bt    = (const float*)d_in[5];
    const float* W1    = (const float*)d_in[6];
    const float* b1    = (const float*)d_in[7];
    const float* W2    = (const float*)d_in[8];
    const float* b2    = (const float*)d_in[9];
    float* out = (float*)d_out;

    // workspace layout (~13.4 MB)
    float* scale = (float*)d_ws;                              // 128 f32
    float* shift = scale + 128;                               // 128 f32
    unsigned short* Hxb = (unsigned short*)(shift + 128);     // 8192*256 bf16 (4MB)
    unsigned short* BTb = Hxb + (size_t)NN * DTH;             // 64*8192 bf16 (1MB)
    unsigned* mwp = (unsigned*)(BTb + (size_t)DH * NN);       // 8192*256 u32 (8MB)

    bn_stats_kernel<<<DIN, 256, 0, stream>>>(H, gamma, beta, scale, shift);
    maskpack_kernel<<<NN, 256, 0, stream>>>(A, mwp);
    prep_kernel<<<NN / 16, 256, 0, stream>>>(H, scale, shift, Wt, bt, W1, b1,
                                             W2, b2, Hxb, BTb, out);
    fused_attn_kernel<<<NN / (16 * 8), 256, 0, stream>>>(Hxb, BTb, mwp, out);
}
